// MultiScale_EEG_Hypergraph_33973191311517
// MI455X (gfx1250) — matrix operand order repaired and checked by reading, not compiled, not run on hardware
//
#include <hip/hip_runtime.h>
#include <hip/hip_bf16.h>

typedef __attribute__((ext_vector_type(16))) __bf16 v16bf;
typedef __attribute__((ext_vector_type(8)))  float  v8f;
typedef __attribute__((ext_vector_type(8)))  unsigned short u16x8;
typedef __attribute__((ext_vector_type(4)))  int    v4i;

// Address-space qualified vector pointers for the async-LDS builtin
// (param types per clang diagnostic: v4i in AS1 (global) / AS3 (LDS)).
typedef __attribute__((address_space(1))) v4i gv4i;
typedef __attribute__((address_space(3))) v4i lv4i;

struct __align__(16) Frag2 { u16x8 a, b; };

constexpr int   kS = 3;
constexpr int   kN = 8192;
constexpr int   kE = 1024;
constexpr int   kD = 512;
constexpr float kAlpha = 3.0f;

// LDS strip: per edge row, 32 k-values of hi (64B) + lo (64B) + 16B pad = 144B.
// Pad => 16-lane ds_load_b128 fragment reads cover all 64 banks exactly once.
constexpr int kRowStrideU16 = 72;              // ushorts per LDS row (144 B)
constexpr int kStripU16     = kE * kRowStrideU16;
constexpr int kMT           = 64;              // rows per workgroup (16 waves)

// ---- CDNA5 async global->LDS staging (ASYNCcnt), with sync fallback --------
#if __has_builtin(__builtin_amdgcn_global_load_async_to_lds_b128)
#define ATHENA_ASYNC_LDS 1
#else
#define ATHENA_ASYNC_LDS 0
#endif

__device__ __forceinline__ void wait_async_all() {
#if ATHENA_ASYNC_LDS
#if __has_builtin(__builtin_amdgcn_s_wait_asynccnt)
    __builtin_amdgcn_s_wait_asynccnt(0);
#else
    asm volatile("s_wait_asynccnt 0x0" ::: "memory");
#endif
#endif
}

// ---------------------------------------------------------------------------
// Pre-pass: split f32 into bf16 hi + bf16 lo planes (a ~= hi + lo, ~2^-17 rel)
// ---------------------------------------------------------------------------
__global__ void convert_split_kernel(const float* __restrict__ src,
                                     unsigned short* __restrict__ hi,
                                     unsigned short* __restrict__ lo,
                                     int n4)
{
    int i = blockIdx.x * blockDim.x + threadIdx.x;
    if (i >= n4) return;
    float4 x = reinterpret_cast<const float4*>(src)[i];
    float f[4] = {x.x, x.y, x.z, x.w};
    unsigned short hs[4], ls[4];
#pragma unroll
    for (int j = 0; j < 4; ++j) {
        __bf16 hb = (__bf16)f[j];
        float  r  = f[j] - (float)hb;
        __bf16 lb = (__bf16)r;
        hs[j] = __builtin_bit_cast(unsigned short, hb);
        ls[j] = __builtin_bit_cast(unsigned short, lb);
    }
    reinterpret_cast<ushort4*>(hi)[i] = make_ushort4(hs[0], hs[1], hs[2], hs[3]);
    reinterpret_cast<ushort4*>(lo)[i] = make_ushort4(ls[0], ls[1], ls[2], ls[3]);
}

// ---------------------------------------------------------------------------
// Stage one B k-strip (all 1024 edges x 32 k, hi+lo) into LDS.
// 512 threads: thread t copies edge rows 2t and 2t+1 (16B chunks).
// Async path: GLOBAL_LOAD_ASYNC_TO_LDS_B128, no VGPR round-trip.
// ---------------------------------------------------------------------------
__device__ __forceinline__ void stage_strip(unsigned short* __restrict__ sbuf,
                                            const unsigned short* __restrict__ BH,
                                            const unsigned short* __restrict__ BL,
                                            int kb, int tid)
{
#pragma unroll
    for (int r = 0; r < 2; ++r) {
        int e = tid * 2 + r;
        const unsigned short* srcH = BH + (size_t)e * kD + kb;
        const unsigned short* srcL = BL + (size_t)e * kD + kb;
        unsigned short* dst = sbuf + e * kRowStrideU16;
#if ATHENA_ASYNC_LDS
#pragma unroll
        for (int j = 0; j < 4; ++j) {
            __builtin_amdgcn_global_load_async_to_lds_b128(
                (gv4i*)(srcH + j * 8), (lv4i*)(dst + j * 8), 0, 0);
            __builtin_amdgcn_global_load_async_to_lds_b128(
                (gv4i*)(srcL + j * 8), (lv4i*)(dst + 32 + j * 8), 0, 0);
        }
#else
        const u16x8* sh = reinterpret_cast<const u16x8*>(srcH);
        const u16x8* sl = reinterpret_cast<const u16x8*>(srcL);
        u16x8* d = reinterpret_cast<u16x8*>(dst);
#pragma unroll
        for (int j = 0; j < 4; ++j) d[j] = sh[j];          // row[0..31]  = hi
#pragma unroll
        for (int j = 0; j < 4; ++j) d[4 + j] = sl[j];      // row[32..63] = lo
#endif
    }
}

// ---------------------------------------------------------------------------
// Main kernel (pre-converted path): 512 thr = 16 waves.
// Row-group g = w>>3 (32 rows each), col chunk cw = w&7 (128 cols each).
// Per wave: 2 row-tiles x 8 col-tiles of 16x16 f32 accumulators.
// ---------------------------------------------------------------------------
__global__ __launch_bounds__(512, 1)
void hgnn_main_kernel(const unsigned short* __restrict__ nodeHi,
                      const unsigned short* __restrict__ nodeLo,
                      const unsigned short* __restrict__ edgeHi,
                      const unsigned short* __restrict__ edgeLo,
                      float* __restrict__ out)
{
    extern __shared__ unsigned short smem[];
    unsigned short* sbuf0 = smem;
    unsigned short* sbuf1 = smem + kStripU16;

    __shared__ float red[8][kMT];
    __shared__ float gm[kMT];
    __shared__ float th[kMT];

    const int s   = blockIdx.y;
    const int nb  = blockIdx.x * kMT;
    const int tid = threadIdx.x;
    const int w   = tid >> 5;
    const int g   = w >> 3;          // row group 0/1
    const int cw  = w & 7;           // col chunk 0..7
    const int l   = tid & 31;
    const int ll  = l & 15;
    const int h   = l >> 4;

    const unsigned short* AH = nodeHi + (size_t)s * kN * kD;
    const unsigned short* AL = nodeLo + (size_t)s * kN * kD;
    const unsigned short* BH = edgeHi + (size_t)s * kE * kD;
    const unsigned short* BL = edgeLo + (size_t)s * kE * kD;

    const v8f vz = {0.f, 0.f, 0.f, 0.f, 0.f, 0.f, 0.f, 0.f};
    v8f acc[2][8];
#pragma unroll
    for (int m = 0; m < 2; ++m)
#pragma unroll
        for (int c = 0; c < 8; ++c) acc[m][c] = vz;

    stage_strip(sbuf0, BH, BL, 0, tid);

    for (int ks = 0; ks < kD / 32; ++ks) {
        const int kb = ks * 32;
        unsigned short* cur = (ks & 1) ? sbuf1 : sbuf0;
        unsigned short* nxt = (ks & 1) ? sbuf0 : sbuf1;

        // cur strip landed; everyone done reading the buffer we overwrite next.
        wait_async_all();
        __syncthreads();
        if (ks + 1 < kD / 32) stage_strip(nxt, BH, BL, kb + 32, tid);

        // A fragments: 16-bit A 16x32 layout. K(elem j) = 8h + (j<8 ? j : 16+j-8)
        v16bf ahi[2], alo[2];
#pragma unroll
        for (int m = 0; m < 2; ++m) {
            const unsigned short* pH =
                AH + (size_t)(nb + g * 32 + m * 16 + ll) * kD + kb + h * 8;
            const unsigned short* pL =
                AL + (size_t)(nb + g * 32 + m * 16 + ll) * kD + kb + h * 8;
            Frag2 fh, fl;
            fh.a = *reinterpret_cast<const u16x8*>(pH);
            fh.b = *reinterpret_cast<const u16x8*>(pH + 16);
            fl.a = *reinterpret_cast<const u16x8*>(pL);
            fl.b = *reinterpret_cast<const u16x8*>(pL + 16);
            ahi[m] = __builtin_bit_cast(v16bf, fh);
            alo[m] = __builtin_bit_cast(v16bf, fl);
        }

#pragma unroll
        for (int c = 0; c < 8; ++c) {
            // B fragment: 16-bit B 32x16 layout. K(elem j) = 8h + (j<8 ? j : 16+j-8)
            const unsigned short* br =
                cur + (size_t)(cw * 128 + c * 16 + ll) * kRowStrideU16 + h * 8;
            Frag2 fh, fl;
            fh.a = *reinterpret_cast<const u16x8*>(br);
            fh.b = *reinterpret_cast<const u16x8*>(br + 16);
            fl.a = *reinterpret_cast<const u16x8*>(br + 32);
            fl.b = *reinterpret_cast<const u16x8*>(br + 48);
            v16bf bhi = __builtin_bit_cast(v16bf, fh);
            v16bf blo = __builtin_bit_cast(v16bf, fl);
#pragma unroll
            for (int m = 0; m < 2; ++m) {
                acc[m][c] = __builtin_amdgcn_wmma_f32_16x16x32_bf16(
                    false, ahi[m], false, bhi, (short)0, acc[m][c], false, false);
                acc[m][c] = __builtin_amdgcn_wmma_f32_16x16x32_bf16(
                    false, ahi[m], false, blo, (short)0, acc[m][c], false, false);
                acc[m][c] = __builtin_amdgcn_wmma_f32_16x16x32_bf16(
                    false, alo[m], false, bhi, (short)0, acc[m][c], false, false);
            }
        }
    }

    // r = relu(alpha * sim)
#pragma unroll
    for (int m = 0; m < 2; ++m)
#pragma unroll
        for (int c = 0; c < 8; ++c)
#pragma unroll
            for (int i = 0; i < 8; ++i)
                acc[m][c][i] = fmaxf(0.f, kAlpha * acc[m][c][i]);

    // C layout: (vgpr i, lane l) -> row g*32 + m*16 + 8h + i, col cw*128 + c*16 + ll
    float pm[2][8];
#pragma unroll
    for (int m = 0; m < 2; ++m)
#pragma unroll
        for (int i = 0; i < 8; ++i) {
            float v = acc[m][0][i];
#pragma unroll
            for (int c = 1; c < 8; ++c) v = fmaxf(v, acc[m][c][i]);
#pragma unroll
            for (int mask = 1; mask < 16; mask <<= 1)
                v = fmaxf(v, __shfl_xor(v, mask, 32));
            pm[m][i] = v;
        }
    if (ll == 0) {
#pragma unroll
        for (int m = 0; m < 2; ++m)
#pragma unroll
            for (int i = 0; i < 8; ++i)
                red[cw][g * 32 + m * 16 + h * 8 + i] = pm[m][i];
    }
    __syncthreads();
    if (tid < kMT) {
        float gmx = red[0][tid];
#pragma unroll
        for (int ww = 1; ww < 8; ++ww) gmx = fmaxf(gmx, red[ww][tid]);
        gm[tid] = gmx;
    }
    __syncthreads();

    float ps[2][8];
#pragma unroll
    for (int m = 0; m < 2; ++m)
#pragma unroll
        for (int i = 0; i < 8; ++i) {
            float gmax = gm[g * 32 + m * 16 + h * 8 + i];
            float v = 0.f;
#pragma unroll
            for (int c = 0; c < 8; ++c) v += __expf(acc[m][c][i] - gmax);
#pragma unroll
            for (int mask = 1; mask < 16; mask <<= 1)
                v += __shfl_xor(v, mask, 32);
            ps[m][i] = v;
        }
    if (ll == 0) {
#pragma unroll
        for (int m = 0; m < 2; ++m)
#pragma unroll
            for (int i = 0; i < 8; ++i)
                red[cw][g * 32 + m * 16 + h * 8 + i] = ps[m][i];
    }
    __syncthreads();
    if (tid < kMT) {
        float z = red[0][tid];
#pragma unroll
        for (int ww = 1; ww < 8; ++ww) z += red[ww][tid];
        th[tid] = gm[tid] + __logf(0.5f * z);   // softmax>0.5 <=> r > max + log(Z/2)
    }
    __syncthreads();

#pragma unroll
    for (int m = 0; m < 2; ++m)
#pragma unroll
        for (int i = 0; i < 8; ++i) {
            int   rg = m * 16 + h * 8 + i;
            float t  = th[g * 32 + rg];
#pragma unroll
            for (int c = 0; c < 8; ++c) {
                out[((size_t)(s * kN + nb + g * 32 + rg)) * kE + cw * 128 + c * 16 + ll] =
                    (acc[m][c][i] > t) ? 1.0f : 0.0f;
            }
        }
}

// ---------------------------------------------------------------------------
// Fallback (ws too small): in-register conversion variant.
// ---------------------------------------------------------------------------
__global__ __launch_bounds__(256, 1)
void hgnn_fallback_kernel(const float* __restrict__ node,
                          const float* __restrict__ edge,
                          float* __restrict__ out)
{
    const int s  = blockIdx.y;
    const int nb = blockIdx.x * 32;
    const int w  = threadIdx.x >> 5;
    const int l  = threadIdx.x & 31;
    const int ll = l & 15;
    const int h  = l >> 4;

    const float* __restrict__ Abase = node + (size_t)s * kN * kD;
    const float* __restrict__ Bbase = edge + (size_t)s * kE * kD;

    const v8f vzero = {0.f, 0.f, 0.f, 0.f, 0.f, 0.f, 0.f, 0.f};
    v8f acc[2][8];
#pragma unroll
    for (int m = 0; m < 2; ++m)
#pragma unroll
        for (int c = 0; c < 8; ++c) acc[m][c] = vzero;

    for (int kb = 0; kb < kD; kb += 32) {
        v16bf ahi[2], alo[2];
#pragma unroll
        for (int m = 0; m < 2; ++m) {
            const float* p = Abase + (size_t)(nb + m * 16 + ll) * kD + kb + h * 8;
            const float4* p4 = reinterpret_cast<const float4*>(p);
            float4 q0 = p4[0], q1 = p4[1], q2 = p4[4], q3 = p4[5];
            float f[16] = {q0.x, q0.y, q0.z, q0.w, q1.x, q1.y, q1.z, q1.w,
                           q2.x, q2.y, q2.z, q2.w, q3.x, q3.y, q3.z, q3.w};
#pragma unroll
            for (int j = 0; j < 16; ++j) {
                __bf16 hi = (__bf16)f[j];
                ahi[m][j] = hi;
                alo[m][j] = (__bf16)(f[j] - (float)hi);
            }
        }
#pragma unroll
        for (int c = 0; c < 8; ++c) {
            const float* p = Bbase + (size_t)(w * 128 + c * 16 + ll) * kD + kb + h * 8;
            const float4* p4 = reinterpret_cast<const float4*>(p);
            float4 q0 = p4[0], q1 = p4[1], q2 = p4[4], q3 = p4[5];
            float f[16] = {q0.x, q0.y, q0.z, q0.w, q1.x, q1.y, q1.z, q1.w,
                           q2.x, q2.y, q2.z, q2.w, q3.x, q3.y, q3.z, q3.w};
            v16bf bhi, blo;
#pragma unroll
            for (int j = 0; j < 16; ++j) {
                __bf16 hi = (__bf16)f[j];
                bhi[j] = hi;
                blo[j] = (__bf16)(f[j] - (float)hi);
            }
#pragma unroll
            for (int m = 0; m < 2; ++m) {
                acc[m][c] = __builtin_amdgcn_wmma_f32_16x16x32_bf16(
                    false, ahi[m], false, bhi, (short)0, acc[m][c], false, false);
                acc[m][c] = __builtin_amdgcn_wmma_f32_16x16x32_bf16(
                    false, ahi[m], false, blo, (short)0, acc[m][c], false, false);
                acc[m][c] = __builtin_amdgcn_wmma_f32_16x16x32_bf16(
                    false, alo[m], false, bhi, (short)0, acc[m][c], false, false);
            }
        }
    }

#pragma unroll
    for (int m = 0; m < 2; ++m)
#pragma unroll
        for (int c = 0; c < 8; ++c)
#pragma unroll
            for (int i = 0; i < 8; ++i)
                acc[m][c][i] = fmaxf(0.f, kAlpha * acc[m][c][i]);

    __shared__ float red[8][32];
    __shared__ float gm[32];
    __shared__ float th[32];

    float pm[2][8];
#pragma unroll
    for (int m = 0; m < 2; ++m)
#pragma unroll
        for (int i = 0; i < 8; ++i) {
            float v = acc[m][0][i];
#pragma unroll
            for (int c = 1; c < 8; ++c) v = fmaxf(v, acc[m][c][i]);
#pragma unroll
            for (int mask = 1; mask < 16; mask <<= 1)
                v = fmaxf(v, __shfl_xor(v, mask, 32));
            pm[m][i] = v;
        }
    if (ll == 0) {
#pragma unroll
        for (int m = 0; m < 2; ++m)
#pragma unroll
            for (int i = 0; i < 8; ++i)
                red[w][m * 16 + h * 8 + i] = pm[m][i];
    }
    __syncthreads();
    if (threadIdx.x < 32) {
        float g2 = red[0][threadIdx.x];
#pragma unroll
        for (int ww = 1; ww < 8; ++ww) g2 = fmaxf(g2, red[ww][threadIdx.x]);
        gm[threadIdx.x] = g2;
    }
    __syncthreads();
    float ps[2][8];
#pragma unroll
    for (int m = 0; m < 2; ++m)
#pragma unroll
        for (int i = 0; i < 8; ++i) {
            float gmax = gm[m * 16 + h * 8 + i];
            float v = 0.f;
#pragma unroll
            for (int c = 0; c < 8; ++c) v += __expf(acc[m][c][i] - gmax);
#pragma unroll
            for (int mask = 1; mask < 16; mask <<= 1)
                v += __shfl_xor(v, mask, 32);
            ps[m][i] = v;
        }
    if (ll == 0) {
#pragma unroll
        for (int m = 0; m < 2; ++m)
#pragma unroll
            for (int i = 0; i < 8; ++i)
                red[w][m * 16 + h * 8 + i] = ps[m][i];
    }
    __syncthreads();
    if (threadIdx.x < 32) {
        float z = red[0][threadIdx.x];
#pragma unroll
        for (int ww = 1; ww < 8; ++ww) z += red[ww][threadIdx.x];
        th[threadIdx.x] = gm[threadIdx.x] + __logf(0.5f * z);
    }
    __syncthreads();

    float* __restrict__ orow = out + ((size_t)s * kN + nb) * kE + w * 128;
#pragma unroll
    for (int m = 0; m < 2; ++m)
#pragma unroll
        for (int i = 0; i < 8; ++i) {
            int   row = m * 16 + h * 8 + i;
            float t   = th[row];
#pragma unroll
            for (int c = 0; c < 8; ++c)
                orow[(size_t)row * kE + c * 16 + ll] =
                    (acc[m][c][i] > t) ? 1.0f : 0.0f;
        }
}

// ---------------------------------------------------------------------------
extern "C" void kernel_launch(void* const* d_in, const int* in_sizes, int n_in,
                              void* d_out, int out_size, void* d_ws, size_t ws_size,
                              hipStream_t stream) {
    (void)in_sizes; (void)n_in; (void)out_size;
    const float* node = (const float*)d_in[1];   // [S,N,D] f32
    const float* edge = (const float*)d_in[2];   // [S,E,D] f32
    float* out = (float*)d_out;                  // [S,N,E] f32

    const size_t nodeCnt = (size_t)kS * kN * kD;
    const size_t edgeCnt = (size_t)kS * kE * kD;
    const size_t need = (nodeCnt + edgeCnt) * 2 * sizeof(unsigned short);

    if (ws_size >= need) {
        unsigned short* nodeHi = (unsigned short*)d_ws;
        unsigned short* nodeLo = nodeHi + nodeCnt;
        unsigned short* edgeHi = nodeLo + nodeCnt;
        unsigned short* edgeLo = edgeHi + edgeCnt;

        int n4n = (int)(nodeCnt / 4);
        int n4e = (int)(edgeCnt / 4);
        convert_split_kernel<<<(n4n + 255) / 256, 256, 0, stream>>>(node, nodeHi, nodeLo, n4n);
        convert_split_kernel<<<(n4e + 255) / 256, 256, 0, stream>>>(edge, edgeHi, edgeLo, n4e);

        size_t shmem = (size_t)2 * kStripU16 * sizeof(unsigned short);  // 288 KB
        dim3 grid(kN / kMT, kS);
        hgnn_main_kernel<<<grid, 512, shmem, stream>>>(nodeHi, nodeLo, edgeHi, edgeLo, out);
    } else {
        dim3 grid(kN / 32, kS);
        hgnn_fallback_kernel<<<grid, 256, 0, stream>>>(node, edge, out);
    }
}
